// ETNNLayer_27822798143665
// MI455X (gfx1250) — matrix operand order repaired and checked by reading, not compiled, not run on hardware
//
#include <hip/hip_runtime.h>
#include <hip/hip_bf16.h>
#include <math.h>

typedef _Float16 h8  __attribute__((ext_vector_type(8)));
typedef _Float16 h16 __attribute__((ext_vector_type(16)));
typedef float    f8  __attribute__((ext_vector_type(8)));

#define HDIM 64

__device__ __forceinline__ float silu_f(float x) {
    return x / (1.0f + __expf(-x));
}

__device__ __forceinline__ f8 wmma_f16(h16 a, h16 b, f8 c) {
    // v_wmma_f32_16x16x32_f16: D = A(16x32 f16) * B(32x16 f16) + C(16x16 f32)
    return __builtin_amdgcn_wmma_f32_16x16x32_f16(false, a, false, b, (short)0, c,
                                                  false, false);
}

// A fragment (16x32 f16) from a row-major f16 LDS tile [16][ldk].
// ISA 7.12.2: lane holds row (lane&15); the 16 halfs are two contiguous runs of 8
// at K = kc*32 + base and kc*32 + base + 16, base = 8*(lane>=16).
__device__ __forceinline__ h16 afrag_lds(const _Float16* base, int lane, int ldk, int kc) {
    int row = lane & 15;
    int off = row * ldk + kc * 32 + ((lane >> 4) & 1) * 8;
    h8 lo = *(const h8*)(base + off);
    h8 hi = *(const h8*)(base + off + 16);
    return __builtin_shufflevector(lo, hi, 0,1,2,3,4,5,6,7,8,9,10,11,12,13,14,15);
}

// B fragment (32x16 f16) from col-major f16 weights W[64][ldk]:
// lane holds column (lane&15)+16n; two runs of 8 at K = kc*32 + base and kc*32 + base + 16,
// base = 8*(lane>=16). Two 16-byte loads per lane.
__device__ __forceinline__ h16 bfrag_gbl(const _Float16* W, int lane, int ldk, int kc, int n) {
    int col = (lane & 15) + 16 * n;
    int k0  = kc * 32 + ((lane >> 4) & 1) * 8;
    h8 lo = *(const h8*)(W + (size_t)col * ldk + k0);
    h8 hi = *(const h8*)(W + (size_t)col * ldk + k0 + 16);
    return __builtin_shufflevector(lo, hi, 0,1,2,3,4,5,6,7,8,9,10,11,12,13,14,15);
}

// ---------------------------------------------------------------------------
// Weight conversion: fp32 row-major (K,64) -> f16 col-major [64][ldk_padded]
// Layout in half-buffer:  W1h[64][160] | W2h[64][64] | P1h[64][64] |
//                         U1h[64][128] | U2h[64][64]
// ---------------------------------------------------------------------------
__global__ void etnn_wconv(const float* __restrict__ W1, const float* __restrict__ W2,
                           const float* __restrict__ P1, const float* __restrict__ U1,
                           const float* __restrict__ U2, _Float16* __restrict__ h) {
    const int nW1 = 64 * 160;     // 10240
    const int nSq = 64 * 64;      // 4096
    const int nU1 = 64 * 128;     // 8192
    int i = blockIdx.x * blockDim.x + threadIdx.x;
    if (i < nW1) {
        int col = i / 160, k = i - col * 160;
        h[i] = (_Float16)(k < 132 ? W1[k * 64 + col] : 0.0f);
    } else if (i < nW1 + nSq) {
        int j = i - nW1; int col = j >> 6, k = j & 63;
        h[i] = (_Float16)W2[k * 64 + col];
    } else if (i < nW1 + 2 * nSq) {
        int j = i - nW1 - nSq; int col = j >> 6, k = j & 63;
        h[i] = (_Float16)P1[k * 64 + col];
    } else if (i < nW1 + 2 * nSq + nU1) {
        int j = i - nW1 - 2 * nSq; int col = j >> 7, k = j & 127;
        h[i] = (_Float16)U1[k * 64 + col];
    } else if (i < nW1 + 3 * nSq + nU1) {
        int j = i - nW1 - 2 * nSq - nU1; int col = j >> 6, k = j & 63;
        h[i] = (_Float16)U2[k * 64 + col];
    }
}

// ---------------------------------------------------------------------------
// Geometry invariants per (batch, edge): [dist, pairwise, centroid, hausdorff]
// ---------------------------------------------------------------------------
__global__ void etnn_geom(const float* __restrict__ positions,
                          const int* __restrict__ edge_index,
                          const int* __restrict__ cni,
                          const float* __restrict__ mask,
                          float* __restrict__ inv4,
                          int Bc, int Cc, int Ec) {
    int gid = blockIdx.x * blockDim.x + threadIdx.x;
    int tot = Bc * Ec;
    if (gid >= tot) return;
    int b = gid / Ec, e = gid - b * Ec;
    int s = edge_index[e], t = edge_index[Ec + e];
    const float* pb = positions + (size_t)b * Cc * 3;

    float r0 = pb[s * 3 + 0] - pb[t * 3 + 0];
    float r1 = pb[s * 3 + 1] - pb[t * 3 + 1];
    float r2 = pb[s * 3 + 2] - pb[t * 3 + 2];
    float dist = sqrtf(r0 * r0 + r1 * r1 + r2 * r2);

    float sp[8][3], dp[8][3], mx[8], my[8], sp2[8], dp2[8];
#pragma unroll
    for (int m = 0; m < 8; ++m) {
        int is = cni[s * 8 + m], it = cni[t * 8 + m];
#pragma unroll
        for (int p = 0; p < 3; ++p) {
            sp[m][p] = pb[is * 3 + p];
            dp[m][p] = pb[it * 3 + p];
        }
        sp2[m] = sp[m][0] * sp[m][0] + sp[m][1] * sp[m][1] + sp[m][2] * sp[m][2];
        dp2[m] = dp[m][0] * dp[m][0] + dp[m][1] * dp[m][1] + dp[m][2] * dp[m][2];
        mx[m] = mask[s * 8 + m];
        my[m] = mask[t * 8 + m];
    }

    const float FINF = __builtin_inff();
    float minxy[8], minyx[8], pairwise = 0.0f;
#pragma unroll
    for (int m = 0; m < 8; ++m) { minxy[m] = FINF; minyx[m] = FINF; }
#pragma unroll
    for (int m = 0; m < 8; ++m) {
#pragma unroll
        for (int n = 0; n < 8; ++n) {
            float cr = sp[m][0] * dp[n][0] + sp[m][1] * dp[n][1] + sp[m][2] * dp[n][2];
            float dd = sp2[m] + dp2[n] - 2.0f * cr;
            dd = dd > 0.0f ? dd : 0.0f;
            float d = sqrtf(dd + 1e-12f);
            pairwise += d * mx[m] * my[n];
            minxy[m] = fminf(minxy[m], my[n] <= 0.0f ? FINF : d);
            minyx[n] = fminf(minyx[n], mx[m] <= 0.0f ? FINF : d);
        }
    }
    float hxy = 0.0f, hyx = 0.0f;
#pragma unroll
    for (int m = 0; m < 8; ++m) {
        hxy = fmaxf(hxy, mx[m] <= 0.0f ? 0.0f : minxy[m]);
        hyx = fmaxf(hyx, my[m] <= 0.0f ? 0.0f : minyx[m]);
    }
    float haus = fmaxf(hxy, hyx);

    float smx = 0.0f, smy = 0.0f;
    float cx0 = 0.0f, cx1 = 0.0f, cx2 = 0.0f, cy0 = 0.0f, cy1 = 0.0f, cy2 = 0.0f;
#pragma unroll
    for (int m = 0; m < 8; ++m) {
        smx += mx[m]; smy += my[m];
        cx0 += sp[m][0] * mx[m]; cx1 += sp[m][1] * mx[m]; cx2 += sp[m][2] * mx[m];
        cy0 += dp[m][0] * my[m]; cy1 += dp[m][1] * my[m]; cy2 += dp[m][2] * my[m];
    }
    float denx = fmaxf(smx, 1e-12f), deny = fmaxf(smy, 1e-12f);
    float g0 = cx0 / denx - cy0 / deny;
    float g1 = cx1 / denx - cy1 / deny;
    float g2 = cx2 / denx - cy2 / deny;
    float centroid = sqrtf(g0 * g0 + g1 * g1 + g2 * g2);

    size_t o = ((size_t)b * Ec + e) * 4;
    inv4[o + 0] = dist;
    inv4[o + 1] = pairwise;
    inv4[o + 2] = centroid;
    inv4[o + 3] = haus;
}

// ---------------------------------------------------------------------------
// Edge message pipeline: msg_in(132) -W1-> silu -W2-> messages(64)
//                        messages -P1-> silu -P2-> tanh -> wts
// Scatter: agg[dst] += messages ; pos_upd[dst] += wts*rel   (fp32 atomics)
// 4 waves/block, 16 edges/wave.
// ---------------------------------------------------------------------------
__global__ void __launch_bounds__(128) etnn_msg(
        const float* __restrict__ features, const float* __restrict__ positions,
        const int* __restrict__ edge_index, const float* __restrict__ inv4,
        const _Float16* __restrict__ gW1h, const _Float16* __restrict__ gW2h,
        const _Float16* __restrict__ gP1h,
        const float* __restrict__ b1, const float* __restrict__ b2,
        const float* __restrict__ pb1, const float* __restrict__ P2,
        const float* __restrict__ pb2,
        float* __restrict__ agg, float* __restrict__ pos_upd,
        int Bc, int Cc, int Ec) {
    __shared__ __align__(32) _Float16 sA[4][16 * 160];   // staging / reshuffle
    __shared__ float sY[4][16 * 64];                     // gate-MLP hidden (f32)
    __shared__ int  ssrc[4][16], sdst[4][16];
    __shared__ float swts[4][16];

    const int lane = threadIdx.x & 31;
    const int w    = threadIdx.x >> 5;
    const int b    = blockIdx.y;
    const int e0   = blockIdx.x * 64 + w * 16;

    // phase 1: edge endpoints
    if (lane < 16) {
        int e  = e0 + lane;
        int ec = e < Ec ? e : Ec - 1;
        ssrc[w][lane] = edge_index[ec];
        sdst[w][lane] = edge_index[Ec + ec];
    }
    __syncthreads();

    // phase 2: stage msg_in = [h_src | h_dst | inv4 | pad] as f16 [16][160]
    for (int idx = lane; idx < 16 * 160; idx += 32) {
        int r = idx / 160, k = idx - r * 160;
        int e = e0 + r; int ec = e < Ec ? e : Ec - 1;
        float v = 0.0f;
        if (k < 64)        v = features[((size_t)b * Cc + ssrc[w][r]) * 64 + k];
        else if (k < 128)  v = features[((size_t)b * Cc + sdst[w][r]) * 64 + (k - 64)];
        else if (k < 132)  v = inv4[((size_t)b * Ec + ec) * 4 + (k - 128)];
        sA[w][idx] = (_Float16)v;
    }
    __syncthreads();

    const int colbase = lane & 15;
    const int mhi = ((lane >> 4) & 1) * 8;

    // GEMM1: (16x160) @ W1h -> 16x64, 5 K-steps x 4 N-tiles
    f8 acc[4] = {};
#pragma unroll
    for (int kc = 0; kc < 5; ++kc) {
        h16 a = afrag_lds(sA[w], lane, 160, kc);
#pragma unroll
        for (int n = 0; n < 4; ++n)
            acc[n] = wmma_f16(a, bfrag_gbl(gW1h, lane, 160, kc, n), acc[n]);
    }
    __syncthreads();

    // bias + silu -> f16 [16][64] in sA
#pragma unroll
    for (int n = 0; n < 4; ++n) {
        int col = colbase + 16 * n;
        float bias = b1[col];
#pragma unroll
        for (int r = 0; r < 8; ++r) {
            float x = silu_f(acc[n][r] + bias);
            sA[w][(r + mhi) * 64 + col] = (_Float16)x;
        }
    }
    __syncthreads();

    // GEMM2: @ W2h -> messages (kept in regs, +b2), also f16 copy into sA
    f8 msg[4] = {};
#pragma unroll
    for (int kc = 0; kc < 2; ++kc) {
        h16 a = afrag_lds(sA[w], lane, 64, kc);
#pragma unroll
        for (int n = 0; n < 4; ++n)
            msg[n] = wmma_f16(a, bfrag_gbl(gW2h, lane, 64, kc, n), msg[n]);
    }
    __syncthreads();
#pragma unroll
    for (int n = 0; n < 4; ++n) {
        int col = colbase + 16 * n;
        float bias = b2[col];
#pragma unroll
        for (int r = 0; r < 8; ++r) {
            float x = msg[n][r] + bias;
            msg[n][r] = x;
            sA[w][(r + mhi) * 64 + col] = (_Float16)x;
        }
    }
    __syncthreads();

    // GEMM3: messages @ P1h, silu -> sY (f32)
    f8 acc3[4] = {};
#pragma unroll
    for (int kc = 0; kc < 2; ++kc) {
        h16 a = afrag_lds(sA[w], lane, 64, kc);
#pragma unroll
        for (int n = 0; n < 4; ++n)
            acc3[n] = wmma_f16(a, bfrag_gbl(gP1h, lane, 64, kc, n), acc3[n]);
    }
#pragma unroll
    for (int n = 0; n < 4; ++n) {
        int col = colbase + 16 * n;
        float bias = pb1[col];
#pragma unroll
        for (int r = 0; r < 8; ++r)
            sY[w][(r + mhi) * 64 + col] = silu_f(acc3[n][r] + bias);
    }
    __syncthreads();

    // wts = tanh(Y @ P2 + pb2), one lane per edge row
    if (lane < 16) {
        float dot = pb2[0];
        for (int k = 0; k < 64; ++k) dot += sY[w][lane * 64 + k] * P2[k];
        swts[w][lane] = tanhf(dot);
    }
    __syncthreads();

    // scatter: agg[b, dst, h] += messages
#pragma unroll
    for (int n = 0; n < 4; ++n) {
        int col = colbase + 16 * n;
#pragma unroll
        for (int r = 0; r < 8; ++r) {
            int el = r + mhi;
            int e  = e0 + el;
            if (e < Ec)
                atomicAdd(&agg[((size_t)b * Cc + sdst[w][el]) * 64 + col], msg[n][r]);
        }
    }
    // scatter: pos_upd[b, dst, :] += wts * rel
    if (lane < 16) {
        int e = e0 + lane;
        if (e < Ec) {
            int s = ssrc[w][lane], t = sdst[w][lane];
            float wt = swts[w][lane];
#pragma unroll
            for (int p = 0; p < 3; ++p) {
                float rel = positions[((size_t)b * Cc + s) * 3 + p]
                          - positions[((size_t)b * Cc + t) * 3 + p];
                atomicAdd(&pos_upd[((size_t)b * Cc + t) * 3 + p], wt * rel);
            }
        }
    }
}

// ---------------------------------------------------------------------------
// Node update: new_features = features + U2(silu(U1([features | agg/deg])))
// 4 waves/block, 16 node-rows/wave; rows are flattened (B*C).
// ---------------------------------------------------------------------------
__global__ void __launch_bounds__(128) etnn_node(
        const float* __restrict__ features, const float* __restrict__ agg,
        const float* __restrict__ degree,
        const _Float16* __restrict__ gU1h, const _Float16* __restrict__ gU2h,
        const float* __restrict__ ub1, const float* __restrict__ ub2,
        float* __restrict__ out_feat, int Bc, int Cc) {
    __shared__ __align__(32) _Float16 sA[4][16 * 128];
    const int lane = threadIdx.x & 31;
    const int w    = threadIdx.x >> 5;
    const int row0 = blockIdx.x * 64 + w * 16;
    const int NR   = Bc * Cc;

    for (int idx = lane; idx < 16 * 128; idx += 32) {
        int r = idx >> 7, k = idx & 127;
        int rho = row0 + r; int rc = rho < NR ? rho : NR - 1;
        float v;
        if (k < 64) {
            v = features[(size_t)rc * 64 + k];
        } else {
            float dg = degree[rc % Cc];
            dg = dg > 1.0f ? dg : 1.0f;
            v = agg[(size_t)rc * 64 + (k - 64)] / dg;
        }
        sA[w][idx] = (_Float16)v;
    }
    __syncthreads();

    const int colbase = lane & 15;
    const int mhi = ((lane >> 4) & 1) * 8;

    f8 acc[4] = {};
#pragma unroll
    for (int kc = 0; kc < 4; ++kc) {
        h16 a = afrag_lds(sA[w], lane, 128, kc);
#pragma unroll
        for (int n = 0; n < 4; ++n)
            acc[n] = wmma_f16(a, bfrag_gbl(gU1h, lane, 128, kc, n), acc[n]);
    }
    __syncthreads();
#pragma unroll
    for (int n = 0; n < 4; ++n) {
        int col = colbase + 16 * n;
        float bias = ub1[col];
#pragma unroll
        for (int r = 0; r < 8; ++r) {
            float x = silu_f(acc[n][r] + bias);
            sA[w][(r + mhi) * 64 + col] = (_Float16)x;
        }
    }
    __syncthreads();

    f8 acc2[4] = {};
#pragma unroll
    for (int kc = 0; kc < 2; ++kc) {
        h16 a = afrag_lds(sA[w], lane, 64, kc);
#pragma unroll
        for (int n = 0; n < 4; ++n)
            acc2[n] = wmma_f16(a, bfrag_gbl(gU2h, lane, 64, kc, n), acc2[n]);
    }
#pragma unroll
    for (int n = 0; n < 4; ++n) {
        int col = colbase + 16 * n;
        float bias = ub2[col];
#pragma unroll
        for (int r = 0; r < 8; ++r) {
            int rho = row0 + r + mhi;
            if (rho < NR)
                out_feat[(size_t)rho * 64 + col] =
                    features[(size_t)rho * 64 + col] + acc2[n][r] + bias;
        }
    }
}

__global__ void etnn_posout(const float* __restrict__ positions,
                            const float* __restrict__ pos_upd,
                            float* __restrict__ out_pos, int n) {
    int i = blockIdx.x * blockDim.x + threadIdx.x;
    if (i < n) out_pos[i] = positions[i] + pos_upd[i];
}

// ---------------------------------------------------------------------------
extern "C" void kernel_launch(void* const* d_in, const int* in_sizes, int n_in,
                              void* d_out, int out_size, void* d_ws, size_t ws_size,
                              hipStream_t stream) {
    (void)n_in; (void)out_size; (void)ws_size;
    const float* features  = (const float*)d_in[0];
    const float* positions = (const float*)d_in[1];
    const int*   edge_index= (const int*)  d_in[2];
    const float* degree    = (const float*)d_in[3];
    const int*   cni       = (const int*)  d_in[4];
    const float* mask      = (const float*)d_in[5];
    const float* W1        = (const float*)d_in[6];
    const float* b1        = (const float*)d_in[7];
    const float* W2        = (const float*)d_in[8];
    const float* b2        = (const float*)d_in[9];
    const float* P1        = (const float*)d_in[10];
    const float* pb1       = (const float*)d_in[11];
    const float* P2        = (const float*)d_in[12];
    const float* pb2       = (const float*)d_in[13];
    const float* U1        = (const float*)d_in[14];
    const float* ub1       = (const float*)d_in[15];
    const float* U2        = (const float*)d_in[16];
    const float* ub2       = (const float*)d_in[17];

    const int Cc = in_sizes[3];
    const int Ec = in_sizes[2] / 2;
    const int Bc = in_sizes[0] / (Cc * HDIM);

    // workspace carve (256B aligned segments)
    char* ws = (char*)d_ws;
    size_t cur = 0;
    auto align256 = [](size_t x) { return (x + 255) & ~(size_t)255; };
    float* agg = (float*)(ws + cur);
    size_t aggBytes = (size_t)Bc * Cc * HDIM * sizeof(float);
    cur = align256(cur + aggBytes);
    float* pos_upd = (float*)(ws + cur);
    size_t puBytes = (size_t)Bc * Cc * 3 * sizeof(float);
    cur = align256(cur + puBytes);
    float* inv4 = (float*)(ws + cur);
    cur = align256(cur + (size_t)Bc * Ec * 4 * sizeof(float));
    _Float16* hw = (_Float16*)(ws + cur);   // 30720 halfs of converted weights

    hipMemsetAsync(agg, 0, aggBytes, stream);
    hipMemsetAsync(pos_upd, 0, puBytes, stream);

    etnn_wconv<<<(30720 + 255) / 256, 256, 0, stream>>>(W1, W2, P1, U1, U2, hw);

    int totGeom = Bc * Ec;
    etnn_geom<<<(totGeom + 255) / 256, 256, 0, stream>>>(
        positions, edge_index, cni, mask, inv4, Bc, Cc, Ec);

    dim3 gmsg((Ec + 63) / 64, Bc);
    etnn_msg<<<gmsg, 128, 0, stream>>>(
        features, positions, edge_index, inv4,
        hw /*W1h*/, hw + 10240 /*W2h*/, hw + 14336 /*P1h*/,
        b1, b2, pb1, P2, pb2, agg, pos_upd, Bc, Cc, Ec);

    int NR = Bc * Cc;
    etnn_node<<<(NR + 63) / 64, 128, 0, stream>>>(
        features, agg, degree, hw + 18432 /*U1h*/, hw + 26624 /*U2h*/,
        ub1, ub2, (float*)d_out, Bc, Cc);

    int npos = Bc * Cc * 3;
    etnn_posout<<<(npos + 255) / 256, 256, 0, stream>>>(
        positions, pos_upd, (float*)d_out + (size_t)NR * HDIM, npos);
}
